// LSTMEncoder_89069031784935
// MI455X (gfx1250) — compile-verified, matrix operand order checked by reading, not run on hardware
//
#include <hip/hip_runtime.h>
#include <hip/hip_bf16.h>

// ---------------------------------------------------------------------------
// LSTM encoder for MI455X (gfx1250, wave32, WMMA).
// Persistent recurrence kernel: 32 blocks x 256 threads, one device-wide
// barrier per timestep, bf16 WMMA (f32 accumulate), cell state in registers.
// Each wave computes 2 M-tiles x 4 gate-tiles so every B fragment feeds two
// WMMAs (halves the L2 weight traffic, the dominant data term: weights are
// L2-resident, x is the only HBM stream and is prefetched one step ahead).
// ---------------------------------------------------------------------------

typedef __bf16 bf16_t;
typedef __attribute__((ext_vector_type(8)))  __bf16 v8bf;
typedef __attribute__((ext_vector_type(16))) __bf16 v16bf;
typedef __attribute__((ext_vector_type(8)))  float  v8f;

#define B_   256
#define T_   512
#define F_   256
#define H_   512
#define E_   128
#define G4_  (4 * H_)     // 2048 gate columns
#define KC_  (F_ + H_)    // 768 concat reduction dim
#define NBLK 32
#define NTHR 256

#define WMMA_BF16(A, Bm, C) \
    __builtin_amdgcn_wmma_f32_16x16x32_bf16(false, (A), false, (Bm), (short)0, (C), false, false)

__device__ __forceinline__ float sigmoidf_(float x) {
    return 1.0f / (1.0f + __expf(-x));
}

// Load one 16x32 bf16 WMMA operand fragment for this lane.
// Per ISA 7.12.2: lane holds row (lane&15); lanes 0-15 cover K pairs
// {0..7, 16..23}, lanes 16-31 cover {8..15, 24..31}. With row-major-over-K
// storage that is two contiguous 8-element (16B) runs at +0 and +32 bytes.
__device__ __forceinline__ v16bf load_frag_g(const bf16_t* p0) {
    v8bf lo = *(const v8bf*)(p0);
    v8bf hi = *(const v8bf*)(p0 + 16);
    return __builtin_shufflevector(lo, hi,
        0, 1, 2, 3, 4, 5, 6, 7, 8, 9, 10, 11, 12, 13, 14, 15);
}

// ---------------------------------------------------------------------------
// Init: pack weights/activations to bf16, fuse biases, seed h ping-pong
// buffer with h0, zero the per-step barrier counters (every launch, so graph
// replays are deterministic).
// ---------------------------------------------------------------------------
__global__ __launch_bounds__(256) void lstm_init_kernel(
    const float* __restrict__ W_ih,  // [2048][256]
    const float* __restrict__ W_hh,  // [2048][512]
    const float* __restrict__ b_ih,  // [2048]
    const float* __restrict__ b_hh,  // [2048]
    const float* __restrict__ W_fc,  // [128][512]
    const float* __restrict__ h0,    // [256][512]
    const float* __restrict__ x,     // [256][512][256]
    bf16_t* __restrict__ Wcat,       // [2048][768]
    float*  __restrict__ bias,       // [2048]
    bf16_t* __restrict__ Wfcb,       // [128][512]
    bf16_t* __restrict__ hbuf,       // [2][256][512], parity 0 seeded
    bf16_t* __restrict__ xbf,        // [256][512][256]
    int*    __restrict__ cnt)        // [512]
{
    const long tid0   = (long)blockIdx.x * blockDim.x + threadIdx.x;
    const long stride = (long)gridDim.x * blockDim.x;

    for (long i = tid0; i < (long)G4_ * KC_; i += stride) {
        int n = (int)(i / KC_), k = (int)(i % KC_);
        float v = (k < F_) ? W_ih[n * F_ + k] : W_hh[n * H_ + (k - F_)];
        Wcat[i] = (bf16_t)v;
    }
    for (long i = tid0; i < G4_; i += stride)
        bias[i] = b_ih[i] + b_hh[i];
    for (long i = tid0; i < (long)E_ * H_; i += stride)
        Wfcb[i] = (bf16_t)W_fc[i];
    for (long i = tid0; i < (long)B_ * H_; i += stride)
        hbuf[i] = (bf16_t)h0[i];
    for (long i = tid0; i < (long)B_ * T_ * F_; i += stride)
        xbf[i] = (bf16_t)x[i];
    for (long i = tid0; i < T_; i += stride)
        cnt[i] = 0;
}

// ---------------------------------------------------------------------------
// Persistent recurrence. Block = (mgroup, jgroup): 32 batch rows x 128 hidden
// cols, all four gate types -> cell update is block-local. Wave w owns hidden
// cols [w*16, w*16+16) for BOTH 16-row M-tiles: 8 accumulators, each B
// fragment reused across the two M-tiles.
// ---------------------------------------------------------------------------
__global__ __launch_bounds__(NTHR) void lstm_persistent_kernel(
    const float*  __restrict__ c0,    // [256][512]
    const bf16_t* __restrict__ xbf,   // [256][512][256]
    const bf16_t* __restrict__ Wcat,  // [2048][768] row-major over K
    const float*  __restrict__ bias,  // [2048]
    bf16_t*       __restrict__ hbuf,  // [2][256][512]
    int*          __restrict__ cnt)   // [512]
{
    const int tid    = threadIdx.x;
    const int lane   = tid & 31;
    const int wave   = tid >> 5;          // 0..7
    const int half   = lane >> 4;         // 0/1 (K-half select)
    const int l15    = lane & 15;
    const int mgroup = blockIdx.x & 7;    // 32-row batch group
    const int jgroup = blockIdx.x >> 3;   // hidden-col group (0..3)

    __shared__ float gates[32][4 * 128];  // [row][gate*128 + jlocal], 64 KB

    // Persistent cell state: 16 (row, j) elements per thread for all steps.
    float c[16];
#pragma unroll
    for (int e = 0; e < 16; ++e) {
        int flat = e * NTHR + tid;        // 0..4095
        int row  = flat >> 7;             // 0..31
        int j    = flat & 127;            // 0..127
        c[e] = c0[(mgroup * 32 + row) * H_ + jgroup * 128 + j];
    }

    const int am0  = mgroup * 32 + l15;        // A row, M-tile 0
    const int am1  = am0 + 16;                 // A row, M-tile 1
    const int jcol = jgroup * 128 + wave * 16 + l15;
    const bf16_t* b0 = Wcat + (long)(0 * H_ + jcol) * KC_;
    const bf16_t* b1 = Wcat + (long)(1 * H_ + jcol) * KC_;
    const bf16_t* b2 = Wcat + (long)(2 * H_ + jcol) * KC_;
    const bf16_t* b3 = Wcat + (long)(3 * H_ + jcol) * KC_;

    for (int t = 0; t < T_; ++t) {
        const int p  = t & 1;
        const int np = p ^ 1;
        const bf16_t* hprev = hbuf + (long)p * (B_ * H_);
        const bf16_t* xr0   = xbf + ((long)am0 * T_ + t) * F_;
        const bf16_t* xr1   = xbf + ((long)am1 * T_ + t) * F_;
        const bf16_t* hr0   = hprev + (long)am0 * H_;
        const bf16_t* hr1   = hprev + (long)am1 * H_;

        // Prefetch next step's x rows (the only HBM-resident stream) so the
        // fetch overlaps this step's h-GEMM + barrier. 32 rows x 512 B.
        if (t + 1 < T_) {
            const bf16_t* pf = xbf +
                ((long)(mgroup * 32 + (tid >> 3)) * T_ + (t + 1)) * F_ +
                (tid & 7) * 32;
            __builtin_prefetch(pf, 0, 3);   // -> global_prefetch_b8
        }

        v8f a00 = {}, a01 = {}, a02 = {}, a03 = {};   // M-tile 0, gates 0..3
        v8f a10 = {}, a11 = {}, a12 = {}, a13 = {};   // M-tile 1, gates 0..3

#define K_STEP(KE, AP0, AP1, KOFF)                                          \
        {                                                                   \
            v16bf fa0 = load_frag_g((AP0) + (KOFF));                        \
            v16bf fa1 = load_frag_g((AP1) + (KOFF));                        \
            v16bf w0  = load_frag_g(b0 + (KE));                             \
            v16bf w1  = load_frag_g(b1 + (KE));                             \
            v16bf w2  = load_frag_g(b2 + (KE));                             \
            v16bf w3  = load_frag_g(b3 + (KE));                             \
            a00 = WMMA_BF16(fa0, w0, a00); a10 = WMMA_BF16(fa1, w0, a10);   \
            a01 = WMMA_BF16(fa0, w1, a01); a11 = WMMA_BF16(fa1, w1, a11);   \
            a02 = WMMA_BF16(fa0, w2, a02); a12 = WMMA_BF16(fa1, w2, a12);   \
            a03 = WMMA_BF16(fa0, w3, a03); a13 = WMMA_BF16(fa1, w3, a13);   \
        }

        // K chunks 0..7: x contribution (K = 0..255)
#pragma unroll 2
        for (int kc = 0; kc < 8; ++kc) {
            const int ke = kc * 32 + half * 8;
            K_STEP(ke, xr0, xr1, ke)
        }
        // K chunks 8..23: h contribution (K = 256..767)
#pragma unroll 2
        for (int kc = 0; kc < 16; ++kc) {
            const int kh = kc * 32 + half * 8;
            const int ke = F_ + kh;
            K_STEP(ke, hr0, hr1, kh)
        }
#undef K_STEP

        // Scatter D tiles (+bias) into LDS per ISA C/D layout:
        // acc[r] = element (M = r + 8*half, N = l15).
#define EMIT_GATE(GI, MT, ACC)                                               \
        {                                                                    \
            float bb = bias[(GI) * H_ + jcol];                               \
            _Pragma("unroll")                                                \
            for (int r = 0; r < 8; ++r)                                      \
                gates[(MT) * 16 + r + half * 8]                              \
                     [(GI) * 128 + wave * 16 + l15] = (ACC)[r] + bb;         \
        }
        EMIT_GATE(0, 0, a00) EMIT_GATE(1, 0, a01)
        EMIT_GATE(2, 0, a02) EMIT_GATE(3, 0, a03)
        EMIT_GATE(0, 1, a10) EMIT_GATE(1, 1, a11)
        EMIT_GATE(2, 1, a12) EMIT_GATE(3, 1, a13)
#undef EMIT_GATE
        __syncthreads();

        // Elementwise LSTM cell update; write h (bf16) to the other parity.
#pragma unroll
        for (int e = 0; e < 16; ++e) {
            int flat = e * NTHR + tid;
            int row  = flat >> 7;
            int j    = flat & 127;
            float iv = sigmoidf_(gates[row][0 * 128 + j]);
            float fv = sigmoidf_(gates[row][1 * 128 + j]);
            float gv = tanhf(gates[row][2 * 128 + j]);
            float ov = sigmoidf_(gates[row][3 * 128 + j]);
            c[e] = fv * c[e] + iv * gv;
            float hv = ov * tanhf(c[e]);
            hbuf[(long)np * (B_ * H_) +
                 (long)(mgroup * 32 + row) * H_ + jgroup * 128 + j] = (bf16_t)hv;
        }

        // Device-wide barrier (one per step; h double-buffering makes one
        // barrier sufficient). Counters are re-zeroed by init each launch.
        __threadfence();
        __syncthreads();
        if (tid == 0) {
            __hip_atomic_fetch_add(&cnt[t], 1, __ATOMIC_RELEASE,
                                   __HIP_MEMORY_SCOPE_AGENT);
            while (__hip_atomic_load(&cnt[t], __ATOMIC_ACQUIRE,
                                     __HIP_MEMORY_SCOPE_AGENT) < NBLK) {
                __builtin_amdgcn_s_sleep(2);
            }
        }
        __syncthreads();
    }
}

// ---------------------------------------------------------------------------
// Final FC: out = sigmoid(hT @ W_fc^T + b_fc), 256x128x512 via WMMA.
// Grid = 16 M-tiles; wave w owns output N-tile w.
// ---------------------------------------------------------------------------
__global__ __launch_bounds__(NTHR) void lstm_fc_kernel(
    const bf16_t* __restrict__ hT,   // hbuf parity 0: [256][512]
    const bf16_t* __restrict__ Wfc,  // [128][512]
    const float*  __restrict__ bfc,  // [128]
    float*        __restrict__ out)  // [256][128]
{
    const int tid   = threadIdx.x;
    const int lane  = tid & 31;
    const int wave  = tid >> 5;
    const int half  = lane >> 4;
    const int l15   = lane & 15;
    const int mtile = blockIdx.x;

    const bf16_t* arow = hT  + (long)(mtile * 16 + l15) * H_;
    const int     n    = wave * 16 + l15;
    const bf16_t* brow = Wfc + (long)n * H_;

    v8f acc = {};
#pragma unroll 4
    for (int kc = 0; kc < 16; ++kc) {
        const int ke = kc * 32 + half * 8;
        v16bf a = load_frag_g(arow + ke);
        v16bf b = load_frag_g(brow + ke);
        acc = WMMA_BF16(a, b, acc);
    }
    const float bb = bfc[n];
#pragma unroll
    for (int r = 0; r < 8; ++r) {
        int row = mtile * 16 + r + half * 8;
        out[row * E_ + n] = sigmoidf_(acc[r] + bb);
    }
}

// ---------------------------------------------------------------------------
extern "C" void kernel_launch(void* const* d_in, const int* in_sizes, int n_in,
                              void* d_out, int out_size, void* d_ws, size_t ws_size,
                              hipStream_t stream)
{
    const float* x    = (const float*)d_in[0];
    const float* h0   = (const float*)d_in[1];
    const float* c0   = (const float*)d_in[2];
    const float* W_ih = (const float*)d_in[3];
    const float* W_hh = (const float*)d_in[4];
    const float* b_ih = (const float*)d_in[5];
    const float* b_hh = (const float*)d_in[6];
    const float* W_fc = (const float*)d_in[7];
    const float* b_fc = (const float*)d_in[8];
    float* out = (float*)d_out;

    // Workspace layout (all offsets 16B-aligned):
    char* ws = (char*)d_ws;
    size_t o = 0;
    bf16_t* Wcat = (bf16_t*)(ws + o); o += (size_t)G4_ * KC_ * sizeof(bf16_t); // 3 MB
    float*  bias = (float*)(ws + o);  o += (size_t)G4_ * sizeof(float);        // 8 KB
    bf16_t* Wfcb = (bf16_t*)(ws + o); o += (size_t)E_ * H_ * sizeof(bf16_t);   // 128 KB
    bf16_t* hbuf = (bf16_t*)(ws + o); o += (size_t)2 * B_ * H_ * sizeof(bf16_t); // 512 KB
    int*    cnt  = (int*)(ws + o);    o += (size_t)T_ * sizeof(int);           // 2 KB
    bf16_t* xbf  = (bf16_t*)(ws + o); o += (size_t)B_ * T_ * F_ * sizeof(bf16_t); // 64 MB

    lstm_init_kernel<<<8192, 256, 0, stream>>>(
        W_ih, W_hh, b_ih, b_hh, W_fc, h0, x, Wcat, bias, Wfcb, hbuf, xbf, cnt);

    lstm_persistent_kernel<<<NBLK, NTHR, 0, stream>>>(
        c0, xbf, Wcat, bias, hbuf, cnt);

    // After t = 511 the final h lands in parity 0.
    lstm_fc_kernel<<<16, NTHR, 0, stream>>>(hbuf, Wfcb, b_fc, out);
}
